// WarpAttn_395136991689
// MI455X (gfx1250) — matrix operand order repaired and checked by reading, not compiled, not run on hardware
//
#include <hip/hip_runtime.h>
#include <math.h>

// ---------------- problem constants ----------------
#define DIMC   1024
#define HEADS  16
#define DHEAD  64
#define MCAM   8
#define PHW    256          // 16*16
#define NTOK   2048         // tokens per side (equi = 32*64, pers = 8*256)
#define FFH    8192
#define FFA    4096
#define NEGV   (-1.0e9f)

typedef _Float16 h8   __attribute__((ext_vector_type(8)));
typedef _Float16 v16h __attribute__((ext_vector_type(16)));
typedef float    v8f  __attribute__((ext_vector_type(8)));

#define WMMA_F16(a,b,c) __builtin_amdgcn_wmma_f32_16x16x32_f16(false,(a),false,(b),(short)0,(c),false,false)

// A-fragment (16x32 f16): caller points p at &A[row][k0 + hi*8]; elements 0..7 = K k0+hi*8..,
// elements 8..15 = K k0+16+hi*8.. per CDNA5 ISA layout.
static __device__ __forceinline__ v16h frag_a(const _Float16* p) {
  h8 lo = *(const h8*)(p);
  h8 hi = *(const h8*)(p + 16);
  return __builtin_shufflevector(lo, hi, 0,1,2,3,4,5,6,7,8,9,10,11,12,13,14,15);
}
// B-fragment (32x16 f16): p at &B[k0+hi*8][col]; elements 0..7 = K k0+hi*8.., 8..15 = K k0+16+hi*8.. (row stride ld).
static __device__ __forceinline__ v16h frag_b(const _Float16* p, size_t ld) {
  v16h r;
  #pragma unroll
  for (int s = 0; s < 8; s++) { r[s] = p[(size_t)s * ld]; r[s + 8] = p[(size_t)(16 + s) * ld]; }
  return r;
}

// ---------------- geometry ----------------
struct Rot { float r00,r01,r02,r10,r11,r12,r20,r21,r22; };
static __device__ __forceinline__ Rot cam_rot(float yaw, float pit) {
  float cy = cosf(yaw), sy = sinf(yaw), cp = cosf(pit), sp = sinf(pit);
  Rot R;          // R = Ry(yaw) @ Rx(pitch)
  R.r00 = cy;  R.r01 = sy*sp;  R.r02 = sy*cp;
  R.r10 = 0.f; R.r11 = cp;     R.r12 = -sp;
  R.r20 = -sy; R.r21 = cy*sp;  R.r22 = cy*cp;
  return R;
}

// map_idx[m][e] = perspective pixel index (iv*16+iu) that equi pixel e lands on for camera m, or -1
__global__ void wa_geo_map(const float* __restrict__ cams, short* __restrict__ mp) {
  int t = blockIdx.x * blockDim.x + threadIdx.x;
  if (t >= MCAM * NTOK) return;
  int m = t >> 11, e = t & 2047;
  float fov = cams[m*3+0], yaw = cams[m*3+1], pit = cams[m*3+2];
  Rot R = cam_rot(yaw, pit);
  int eh = e >> 6, ew = e & 63;
  float th = (ew + 0.5f) * (6.28318530717958647692f / 64.f) - 3.14159265358979323846f;
  float ph = 1.57079632679489661923f - (eh + 0.5f) * (3.14159265358979323846f / 32.f);
  float cph = cosf(ph);
  float dx = cph * sinf(th), dy = sinf(ph), dz = cph * cosf(th);
  // dc = R^T d
  float x = R.r00*dx + R.r10*dy + R.r20*dz;
  float y = R.r01*dx + R.r11*dy + R.r21*dz;
  float z = R.r02*dx + R.r12*dy + R.r22*dz;
  bool valid = z > 1e-6f;
  float zs = valid ? z : 1.f;
  float f = 8.0f / tanf(0.5f * fov);
  float u = f * x / zs + 7.5f;
  float v = -f * y / zs + 7.5f;
  float iu = rintf(u), iv = rintf(v);
  bool inb = valid && iu >= 0.f && iu < 16.f && iv >= 0.f && iv < 16.f;
  mp[t] = inb ? (short)((int)iv * 16 + (int)iu) : (short)(-1);
}

__global__ void wa_pers_coords(const float* __restrict__ cams, float* __restrict__ co) {
  int t = blockIdx.x * blockDim.x + threadIdx.x;
  if (t >= MCAM * PHW) return;
  int m = t >> 8, pix = t & 255;
  int i = pix >> 4, j = pix & 15;
  float fov = cams[m*3+0];
  Rot R = cam_rot(cams[m*3+1], cams[m*3+2]);
  float f = 8.0f / tanf(0.5f * fov);
  float x = (j - 7.5f) / f, y = -(i - 7.5f) / f;
  float n = rsqrtf(x*x + y*y + 1.f);
  float dx = x*n, dy = y*n, dz = n;
  float wx = R.r00*dx + R.r01*dy + R.r02*dz;
  float wy = R.r10*dx + R.r11*dy + R.r12*dz;
  float wz = R.r20*dx + R.r21*dy + R.r22*dz;
  co[t*2+0] = atan2f(wx, wz);
  co[t*2+1] = asinf(fminf(1.f, fmaxf(-1.f, wy)));
}

__global__ void wa_equi_coords(float* __restrict__ co) {
  int e = blockIdx.x * blockDim.x + threadIdx.x;
  if (e >= NTOK) return;
  int eh = e >> 6, ew = e & 63;
  co[e*2+0] = (ew + 0.5f) * (6.28318530717958647692f / 64.f) - 3.14159265358979323846f;
  co[e*2+1] = 1.57079632679489661923f - (eh + 0.5f) * (3.14159265358979323846f / 32.f);
}

// spherical PE: channel 4*band+{sin th, sin ph, cos th, cos ph}, freq = band+1
__global__ void wa_spe(const float* __restrict__ co, float* __restrict__ pe) {
  int t = blockIdx.x * blockDim.x + threadIdx.x;
  if (t >= NTOK * 256) return;
  int tok = t >> 8, band = t & 255;
  float fr = (float)(band + 1);
  float at = co[tok*2+0] * fr, ap = co[tok*2+1] * fr;
  float* o = pe + (size_t)tok * DIMC + band * 4;
  o[0] = sinf(at); o[1] = sinf(ap); o[2] = cosf(at); o[3] = cosf(ap);
}

// ---------------- layout / elementwise ----------------
__global__ void wa_chw_to_tok(const float* __restrict__ x, float* __restrict__ o, int hw, int total) {
  int t = blockIdx.x * blockDim.x + threadIdx.x;
  if (t >= total) return;
  int row = t >> 10, c = t & 1023;
  int img = row / hw, pix = row - img * hw;
  o[t] = x[((size_t)img * DIMC + c) * hw + pix];
}

__global__ void wa_f32_to_f16(const float* __restrict__ a, _Float16* __restrict__ o, int n) {
  int t = blockIdx.x * blockDim.x + threadIdx.x;
  if (t < n) o[t] = (_Float16)a[t];
}

__global__ void wa_add_pe(const float* __restrict__ x, const float* __restrict__ pe,
                          _Float16* __restrict__ o, int n) {
  int t = blockIdx.x * blockDim.x + threadIdx.x;
  if (t < n) o[t] = (_Float16)(x[t] + pe[t]);
}

__global__ __launch_bounds__(128)
void wa_ln(const float* __restrict__ X, const float* __restrict__ g, const float* __restrict__ b,
           const float* __restrict__ pe, _Float16* __restrict__ Y) {
  int row = blockIdx.x;
  const float* x = X + (size_t)row * DIMC;
  float s = 0.f, s2 = 0.f;
  for (int c = threadIdx.x; c < DIMC; c += 128) { float v = x[c]; s += v; s2 += v * v; }
  for (int m = 16; m >= 1; m >>= 1) { s += __shfl_xor(s, m, 32); s2 += __shfl_xor(s2, m, 32); }
  __shared__ float ws[4], ws2[4];
  int w = threadIdx.x >> 5;
  if ((threadIdx.x & 31) == 0) { ws[w] = s; ws2[w] = s2; }
  __syncthreads();
  s = ws[0] + ws[1] + ws[2] + ws[3];
  s2 = ws2[0] + ws2[1] + ws2[2] + ws2[3];
  float mu = s * (1.f / DIMC);
  float var = s2 * (1.f / DIMC) - mu * mu;
  float rstd = rsqrtf(var + 1e-5f);
  for (int c = threadIdx.x; c < DIMC; c += 128) {
    float v = (x[c] - mu) * rstd * g[c] + b[c];
    if (pe) v += pe[(size_t)row * DIMC + c];
    Y[(size_t)row * DIMC + c] = (_Float16)v;
  }
}

__global__ void wa_geglu(const _Float16* __restrict__ ug, _Float16* __restrict__ act, int n) {
  int t = blockIdx.x * blockDim.x + threadIdx.x;
  if (t >= n) return;
  int r = t >> 12, c = t & 4095;
  float u = (float)ug[(size_t)r * FFH + c];
  float gg = (float)ug[(size_t)r * FFH + FFA + c];
  float ge = 0.5f * gg * (1.f + erff(gg * 0.70710678118654752440f));
  act[t] = (_Float16)(u * ge);
}

__global__ void wa_out_pers(const float* __restrict__ y, float* __restrict__ o) {
  int t = blockIdx.x * blockDim.x + threadIdx.x;
  if (t >= MCAM * DIMC * PHW) return;
  int m = t / (DIMC * PHW);
  int rem = t - m * (DIMC * PHW);
  int c = rem >> 8, pix = rem & 255;
  o[t] = y[(size_t)(m * PHW + pix) * DIMC + c];
}

__global__ void wa_out_equi(const float* __restrict__ y, float* __restrict__ o) {
  int t = blockIdx.x * blockDim.x + threadIdx.x;
  if (t >= DIMC * NTOK) return;
  int c = t >> 11, e = t & 2047;
  o[t] = y[(size_t)e * DIMC + c];
}

// ---------------- WMMA GEMM: C[Mr x N] = A[Mr x K](f16) * B[K x N](f16) ----------------
// Wave tile: 32 rows x 64 cols (2 A-fragments x 4 B-fragments -> 8 WMMAs per K-step;
// every B-fragment is reused for two row-groups to halve B traffic per FLOP).
// B pointer is a carried induction pointer (bp += 32*N per K-step) so address gen is a
// single add instead of a per-iteration 64-bit multiply.
// mode bit0: f16 out; bit1: transposed f16 out (Ch[col*Mr+row]); bit2: +bias[col]; bit3: +resid[row*N+col]
__global__ __launch_bounds__(256)
void wa_gemm(const _Float16* __restrict__ A, const _Float16* __restrict__ B,
             float* __restrict__ Cf, _Float16* __restrict__ Ch,
             const float* __restrict__ bias, const float* __restrict__ resid,
             int Mr, int N, int K, int mode) {
  int lane = threadIdx.x & 31;
  int wid  = (blockIdx.x * blockDim.x + threadIdx.x) >> 5;
  int tilesN = N >> 6;
  int tiles  = (Mr >> 5) * tilesN;
  if (wid >= tiles) return;
  int tm = wid / tilesN, tn = wid - tm * tilesN;
  int hi = lane >> 4, ln15 = lane & 15;
  int n0 = tn * 64;
  const _Float16* ap0 = A + (size_t)(tm * 32 + ln15) * K + hi * 8;        // rows 0..15 of tile
  const _Float16* ap1 = A + (size_t)(tm * 32 + 16 + ln15) * K + hi * 8;   // rows 16..31
  const _Float16* bp  = B + (size_t)(hi * 8) * N + n0 + ln15;             // col = ln15, carried
  size_t bstep = (size_t)32 * N;
  v8f zero = {0.f,0.f,0.f,0.f,0.f,0.f,0.f,0.f};
  v8f acc[2][4] = {{zero, zero, zero, zero}, {zero, zero, zero, zero}};
  for (int k0 = 0; k0 < K; k0 += 32) {
    v16h a0 = frag_a(ap0 + k0);
    v16h a1 = frag_a(ap1 + k0);
    v16h b0 = frag_b(bp, N);
    v16h b1 = frag_b(bp + 16, N);
    v16h b2 = frag_b(bp + 32, N);
    v16h b3 = frag_b(bp + 48, N);
    bp += bstep;
    acc[0][0] = WMMA_F16(a0, b0, acc[0][0]);
    acc[1][0] = WMMA_F16(a1, b0, acc[1][0]);
    acc[0][1] = WMMA_F16(a0, b1, acc[0][1]);
    acc[1][1] = WMMA_F16(a1, b1, acc[1][1]);
    acc[0][2] = WMMA_F16(a0, b2, acc[0][2]);
    acc[1][2] = WMMA_F16(a1, b2, acc[1][2]);
    acc[0][3] = WMMA_F16(a0, b3, acc[0][3]);
    acc[1][3] = WMMA_F16(a1, b3, acc[1][3]);
  }
  #pragma unroll
  for (int g = 0; g < 2; g++) {
    #pragma unroll
    for (int nt = 0; nt < 4; nt++) {
      int col = n0 + nt * 16 + ln15;
      float bv = (mode & 4) ? bias[col] : 0.f;
      if (mode & 2) {
        h8 pk;
        #pragma unroll
        for (int v = 0; v < 8; v++) pk[v] = (_Float16)(acc[g][nt][v] + bv);
        *(h8*)(Ch + (size_t)col * Mr + tm * 32 + g * 16 + 8 * hi) = pk;
      } else {
        #pragma unroll
        for (int v = 0; v < 8; v++) {
          int r = tm * 32 + g * 16 + v + 8 * hi;
          float val = acc[g][nt][v] + bv;
          if (mode & 8) val += resid[(size_t)r * N + col];
          if (mode & 1) Ch[(size_t)r * N + col] = (_Float16)val;
          else          Cf[(size_t)r * N + col] = val;
        }
      }
    }
  }
}

// ---------------- flash attention (masked, streaming softmax) ----------------
// Q: [NTOK x DIMC] f16 (row-major).  Kt: [DIMC x NTOK] f16 (d-major).  V: [NTOK x DIMC] f16.
// dir==0: query = equi pixel e, key index = m*256+p.  dir==1: query = (m,p), key = e.
__global__ __launch_bounds__(256)
void wa_attn(const _Float16* __restrict__ Q, const _Float16* __restrict__ Kt,
             const _Float16* __restrict__ V, const short* __restrict__ mp,
             _Float16* __restrict__ O, int dir) {
  __shared__ __align__(16) _Float16 plds[8][16 * 32];
  int lane = threadIdx.x & 31;
  int wl   = threadIdx.x >> 5;
  int wg   = blockIdx.x * 8 + wl;
  int head = wg >> 7;
  int qt   = wg & 127;
  int hi = lane >> 4, ln15 = lane & 15;

  const _Float16* qp = Q + (size_t)(qt * 16 + ln15) * DIMC + head * DHEAD + hi * 8;
  v16h qa0 = frag_a(qp);
  v16h qa1 = frag_a(qp + 32);

  v8f zero = {0.f,0.f,0.f,0.f,0.f,0.f,0.f,0.f};
  v8f acc[4] = {zero, zero, zero, zero};
  float rm[8], rs[8];
  #pragma unroll
  for (int v = 0; v < 8; v++) { rm[v] = -3.0e38f; rs[v] = 0.f; }

  for (int kb = 0; kb < NTOK; kb += 32) {
    const _Float16* kp0 = Kt + (size_t)(head * DHEAD + hi * 8) * NTOK + kb + ln15;
    const _Float16* kp1 = Kt + (size_t)(head * DHEAD + 32 + hi * 8) * NTOK + kb + ln15;
    v16h b00 = frag_b(kp0, NTOK);        // keys kb..kb+15, d 0..31
    v16h b01 = frag_b(kp0 + 16, NTOK);   // keys kb+16..kb+31
    v16h b10 = frag_b(kp1, NTOK);        // d 32..63
    v16h b11 = frag_b(kp1 + 16, NTOK);
    v8f s0 = WMMA_F16(qa0, b00, zero); s0 = WMMA_F16(qa1, b10, s0);
    v8f s1 = WMMA_F16(qa0, b01, zero); s1 = WMMA_F16(qa1, b11, s1);

    #pragma unroll
    for (int v = 0; v < 8; v++) {
      int q = qt * 16 + v + 8 * hi;
      int k0i = kb + ln15, k1i = kb + 16 + ln15;
      int m0, p0, e0, m1, p1, e1;
      if (dir == 0) { e0 = q; e1 = q; m0 = k0i >> 8; p0 = k0i & 255; m1 = k1i >> 8; p1 = k1i & 255; }
      else          { m0 = q >> 8; p0 = q & 255; m1 = m0; p1 = p0; e0 = k0i; e1 = k1i; }
      s0[v] = ((int)mp[m0 * NTOK + e0] == p0) ? s0[v] * 0.125f : NEGV;
      s1[v] = ((int)mp[m1 * NTOK + e1] == p1) ? s1[v] * 0.125f : NEGV;
      // online softmax (row spans 16 lanes of one half-wave)
      float t = fmaxf(s0[v], s1[v]);
      t = fmaxf(t, __shfl_xor(t, 1, 32));
      t = fmaxf(t, __shfl_xor(t, 2, 32));
      t = fmaxf(t, __shfl_xor(t, 4, 32));
      t = fmaxf(t, __shfl_xor(t, 8, 32));
      float nm = fmaxf(rm[v], t);
      float corr = __expf(rm[v] - nm);
      float p0e = __expf(s0[v] - nm);
      float p1e = __expf(s1[v] - nm);
      float ssum = p0e + p1e;
      ssum += __shfl_xor(ssum, 1, 32);
      ssum += __shfl_xor(ssum, 2, 32);
      ssum += __shfl_xor(ssum, 4, 32);
      ssum += __shfl_xor(ssum, 8, 32);
      rs[v] = rs[v] * corr + ssum;
      rm[v] = nm;
      #pragma unroll
      for (int nt = 0; nt < 4; nt++) acc[nt][v] *= corr;
      int qloc = v + 8 * hi;
      plds[wl][qloc * 32 + ln15]      = (_Float16)p0e;
      plds[wl][qloc * 32 + 16 + ln15] = (_Float16)p1e;
    }
    // same-wave DS RAW: re-stripe P from C-layout to A-layout through LDS
    asm volatile("s_wait_dscnt 0" ::: "memory");
    v16h pa = frag_a(&plds[wl][ln15 * 32 + hi * 8]);
    #pragma unroll
    for (int nt = 0; nt < 4; nt++) {
      const _Float16* vp = V + (size_t)(kb + hi * 8) * DIMC + head * DHEAD + nt * 16 + ln15;
      v16h vb = frag_b(vp, DIMC);
      acc[nt] = WMMA_F16(pa, vb, acc[nt]);
    }
  }
  #pragma unroll
  for (int nt = 0; nt < 4; nt++)
    #pragma unroll
    for (int v = 0; v < 8; v++) {
      int r = qt * 16 + v + 8 * hi;
      O[(size_t)r * DIMC + head * DHEAD + nt * 16 + ln15] = (_Float16)(acc[nt][v] / rs[v]);
    }
}

// ---------------- host driver ----------------
static inline int cdiv(long a, long b) { return (int)((a + b - 1) / b); }

extern "C" void kernel_launch(void* const* d_in, const int* in_sizes, int n_in,
                              void* d_out, int out_size, void* d_ws, size_t ws_size,
                              hipStream_t stream) {
  (void)in_sizes; (void)n_in; (void)out_size; (void)ws_size;
  const float* pers_x = (const float*)d_in[0];
  const float* equi_x = (const float*)d_in[1];
  const float* cams   = (const float*)d_in[2];
  const float* ln1g = (const float*)d_in[3];
  const float* ln1b = (const float*)d_in[4];
  const float* ln2g = (const float*)d_in[5];
  const float* ln2b = (const float*)d_in[6];
  const float* Wq = (const float*)d_in[7];
  const float* Wk = (const float*)d_in[8];
  const float* Wv = (const float*)d_in[9];
  const float* Wo = (const float*)d_in[10];
  const float* bo = (const float*)d_in[11];
  const float* W1 = (const float*)d_in[12];
  const float* b1 = (const float*)d_in[13];
  const float* W2 = (const float*)d_in[14];
  const float* b2 = (const float*)d_in[15];
  float* out = (float*)d_out;

  char* wsb = (char*)d_ws;
  size_t off = 0;
  auto take = [&](size_t bytes) -> char* {
    char* p = wsb + off; off += (bytes + 255) & ~(size_t)255; return p;
  };
  short*    mapidx = (short*)   take((size_t)MCAM * NTOK * 2);
  float*    coordP = (float*)   take((size_t)NTOK * 2 * 4);
  float*    coordE = (float*)   take((size_t)NTOK * 2 * 4);
  float*    peP    = (float*)   take((size_t)NTOK * DIMC * 4);
  float*    peE    = (float*)   take((size_t)NTOK * DIMC * 4);
  float*    xinP   = (float*)   take((size_t)NTOK * DIMC * 4);
  float*    xinE   = (float*)   take((size_t)NTOK * DIMC * 4);
  _Float16* wq16   = (_Float16*)take((size_t)DIMC * DIMC * 2);
  _Float16* wk16   = (_Float16*)take((size_t)DIMC * DIMC * 2);
  _Float16* wv16   = (_Float16*)take((size_t)DIMC * DIMC * 2);
  _Float16* wo16   = (_Float16*)take((size_t)DIMC * DIMC * 2);
  _Float16* w116   = (_Float16*)take((size_t)DIMC * FFH * 2);
  _Float16* w216   = (_Float16*)take((size_t)FFA * DIMC * 2);
  _Float16* hq     = (_Float16*)take((size_t)NTOK * DIMC * 2);
  _Float16* ctx    = (_Float16*)take((size_t)NTOK * DIMC * 2);
  _Float16* qh     = (_Float16*)take((size_t)NTOK * DIMC * 2);
  _Float16* kt     = (_Float16*)take((size_t)DIMC * NTOK * 2);
  _Float16* vh     = (_Float16*)take((size_t)NTOK * DIMC * 2);
  _Float16* ao     = (_Float16*)take((size_t)NTOK * DIMC * 2);
  float*    x2     = (float*)   take((size_t)NTOK * DIMC * 4);
  _Float16* hh2    = (_Float16*)take((size_t)NTOK * DIMC * 2);
  _Float16* ffug   = (_Float16*)take((size_t)NTOK * FFH * 2);
  _Float16* act    = (_Float16*)take((size_t)NTOK * FFA * 2);
  float*    yout   = (float*)   take((size_t)NTOK * DIMC * 4);

  const int TB = 256;
  // geometry + positional encodings
  wa_geo_map    <<<cdiv(MCAM*NTOK, TB), TB, 0, stream>>>(cams, mapidx);
  wa_pers_coords<<<cdiv(MCAM*PHW, TB), TB, 0, stream>>>(cams, coordP);
  wa_equi_coords<<<cdiv(NTOK, TB), TB, 0, stream>>>(coordE);
  wa_spe<<<cdiv((long)NTOK*256, TB), TB, 0, stream>>>(coordP, peP);
  wa_spe<<<cdiv((long)NTOK*256, TB), TB, 0, stream>>>(coordE, peE);
  // channel-major -> token-major
  wa_chw_to_tok<<<cdiv((long)NTOK*DIMC, TB), TB, 0, stream>>>(pers_x, xinP, PHW, NTOK*DIMC);
  wa_chw_to_tok<<<cdiv((long)NTOK*DIMC, TB), TB, 0, stream>>>(equi_x, xinE, NTOK, NTOK*DIMC);
  // weights -> f16
  wa_f32_to_f16<<<cdiv((long)DIMC*DIMC, TB), TB, 0, stream>>>(Wq, wq16, DIMC*DIMC);
  wa_f32_to_f16<<<cdiv((long)DIMC*DIMC, TB), TB, 0, stream>>>(Wk, wk16, DIMC*DIMC);
  wa_f32_to_f16<<<cdiv((long)DIMC*DIMC, TB), TB, 0, stream>>>(Wv, wv16, DIMC*DIMC);
  wa_f32_to_f16<<<cdiv((long)DIMC*DIMC, TB), TB, 0, stream>>>(Wo, wo16, DIMC*DIMC);
  wa_f32_to_f16<<<cdiv((long)DIMC*FFH, TB), TB, 0, stream>>>(W1, w116, DIMC*FFH);
  wa_f32_to_f16<<<cdiv((long)FFA*DIMC, TB), TB, 0, stream>>>(W2, w216, FFA*DIMC);

  auto gemm = [&](const _Float16* A, const _Float16* B, int Mr, int N, int K, int mode,
                  const float* bias, const float* resid, float* Cf, _Float16* Ch) {
    long waves = ((long)Mr / 32) * (N / 64);   // 32x64 tile per wave
    wa_gemm<<<cdiv(waves * 32, TB), TB, 0, stream>>>(A, B, Cf, Ch, bias, resid, Mr, N, K, mode);
  };

  auto run_dir = [&](const float* xq, const float* xkv, const float* peq, const float* pekv, int dir) {
    wa_ln<<<NTOK, 128, 0, stream>>>(xq, ln1g, ln1b, peq, hq);                 // h = LN(x)+qpe
    wa_add_pe<<<cdiv((long)NTOK*DIMC, TB), TB, 0, stream>>>(xkv, pekv, ctx, NTOK*DIMC);
    gemm(hq,  wq16, NTOK, DIMC, DIMC, 1,  nullptr, nullptr, nullptr, qh);    // Q
    gemm(ctx, wk16, NTOK, DIMC, DIMC, 2,  nullptr, nullptr, nullptr, kt);    // K (transposed out)
    gemm(ctx, wv16, NTOK, DIMC, DIMC, 1,  nullptr, nullptr, nullptr, vh);    // V
    wa_attn<<<(HEADS*128)/8, TB, 0, stream>>>(qh, kt, vh, mapidx, ao, dir);
    gemm(ao,  wo16, NTOK, DIMC, DIMC, 12, bo, xq, x2, nullptr);              // x2 = x + O@Wo + bo
    wa_ln<<<NTOK, 128, 0, stream>>>(x2, ln2g, ln2b, nullptr, hh2);
    gemm(hh2, w116, NTOK, FFH, DIMC, 5,  b1, nullptr, nullptr, ffug);        // GEGLU up
    wa_geglu<<<cdiv((long)NTOK*FFA, TB), TB, 0, stream>>>(ffug, act, NTOK*FFA);
    gemm(act, w216, NTOK, DIMC, FFA, 12, b2, x2, yout, nullptr);             // out = x2 + act@W2 + b2
  };

  // direction 1: equi queries over perspective kv -> equi_out (second half of d_out)
  run_dir(xinE, xinP, peE, peP, 0);
  wa_out_equi<<<cdiv((long)DIMC*NTOK, TB), TB, 0, stream>>>(yout, out + (size_t)MCAM*DIMC*PHW);
  // direction 2: perspective queries over equi kv -> pers_out (first half of d_out)
  run_dir(xinP, xinE, peP, peE, 1);
  wa_out_pers<<<cdiv((long)MCAM*DIMC*PHW, TB), TB, 0, stream>>>(yout, out);
}
